// GodheadTransformer_74792560492641
// MI455X (gfx1250) — matrix operand order repaired and checked by reading, not compiled, not run on hardware
//
#include <hip/hip_runtime.h>

// GodheadTransformer forward for MI455X (gfx1250, wave32, WMMA + TDM).
// All GEMMs (QKV, proj, thought MLP, MoE experts) and the windowed attention
// run through v_wmma_f32_16x16x32_f16 with f32 accumulation. GEMM tiles are
// staged to LDS by the Tensor Data Mover (tensor_load_to_lds + s_wait_tensorcnt)
// when the builtin is available, with a ds_store_b128 fallback.

typedef __attribute__((ext_vector_type(16))) _Float16 v16h;
typedef __attribute__((ext_vector_type(8)))  float    v8f;
typedef __attribute__((ext_vector_type(4)))  unsigned int u32x4;
typedef __attribute__((ext_vector_type(8)))  int      i32x8;
typedef __attribute__((ext_vector_type(4)))  int      i32x4;

#define LAYERS 6
#define C      384
#define H      6
#define HD     64
#define T      256
#define BATCH  32
#define E      4
#define DT     256
#define WWIN   64
#define NROW   (BATCH * T)   // 8192 tokens
#define C3     (3 * C)       // 1152
#define FF     (4 * C)       // 1536
#define SCL    0.125f        // HD^-0.5

// LDS tile: 64 rows x 32 k-dwords, padded +4 dwords per row (bank spread).
#define TROWS   64
#define TK      32
#define TPAD    4
#define TSTRIDE (TK + TPAD)                       // 36 floats
#define TILEBYTES (TROWS * TSTRIDE * 4)           // 9216 B

#define CVT4(vec, idx, f4)                      \
  (vec)[(idx) + 0] = (_Float16)(f4).x;          \
  (vec)[(idx) + 1] = (_Float16)(f4).y;          \
  (vec)[(idx) + 2] = (_Float16)(f4).z;          \
  (vec)[(idx) + 3] = (_Float16)(f4).w;

#if defined(__has_builtin)
#if __has_builtin(__builtin_amdgcn_tensor_load_to_lds) && \
    __has_builtin(__builtin_amdgcn_s_wait_tensorcnt)
#define USE_TDM 1
#endif
#endif
#ifndef USE_TDM
#define USE_TDM 0
#endif

__device__ __forceinline__ float wredsum(float v) {
  for (int off = 16; off >= 1; off >>= 1) v += __shfl_xor(v, off, 32);
  return v;
}

// Generic pointers to LDS carry the byte offset in addr[31:0] (ISA aperture).
__device__ __forceinline__ unsigned lds_off(const void* p) {
  return (unsigned)(unsigned long long)p;
}

#if USE_TDM
// Issue one TDM 2D tile load: 32 x 64 elements of f32, row stride K, into LDS
// with 4-dword padding every 32 dwords (-> TSTRIDE float rows in LDS).
__device__ __forceinline__ void tdm_load_tile_f32(const float* gptr,
                                                  unsigned ldsByteOff,
                                                  int rowStride, int d0, int d1) {
  unsigned long long ga = (unsigned long long)gptr;
  u32x4 g0;
  g0.x = 1u;                                        // count=1, load, not restore
  g0.y = ldsByteOff;                                // lds_addr
  g0.z = (unsigned)(ga & 0xFFFFFFFFu);              // global_addr[31:0]
  g0.w = (unsigned)((ga >> 32) & 0x1FFFFFFu) | (2u << 30);  // addr[56:32]|type=2
  i32x8 g1;
  // data_size=4B(2), pad_enable, pad_interval=32dw(4), pad_amount=4dw(3)
  g1[0] = (int)((2u << 16) | (1u << 20) | (4u << 22) | (3u << 25));
  g1[1] = (int)(((unsigned)d0 & 0xFFFFu) << 16);            // tensor_dim0 lo16
  g1[2] = (int)((((unsigned)d0 >> 16) & 0xFFFFu) |
                (((unsigned)d1 & 0xFFFFu) << 16));          // dim0 hi | dim1 lo
  g1[3] = (int)((((unsigned)d1 >> 16) & 0xFFFFu) |
                ((unsigned)TK << 16));                      // dim1 hi | tile_dim0
  g1[4] = (int)TROWS;                                       // tile_dim1 | tile_dim2=0
  g1[5] = rowStride;                                        // tensor_dim0_stride lo32
  g1[6] = 0;
  g1[7] = 0;
  i32x4 z4 = {0, 0, 0, 0};
#if defined(__clang_major__) && (__clang_major__ >= 23)
  i32x8 z8 = {0, 0, 0, 0, 0, 0, 0, 0};
  __builtin_amdgcn_tensor_load_to_lds(g0, g1, z4, z4, z8, 0);
#else
  __builtin_amdgcn_tensor_load_to_lds(g0, g1, z4, z4, 0);
#endif
}
#endif

// ---------------------------------------------------------------------------
// LayerNorm: one wave per token row.
// ---------------------------------------------------------------------------
__global__ void ln_rows(const float* __restrict__ x, const float* __restrict__ g,
                        const float* __restrict__ b, float* __restrict__ o) {
  int wid = threadIdx.x >> 5, lane = threadIdx.x & 31;
  int row = blockIdx.x * (blockDim.x >> 5) + wid;
  if (row >= NROW) return;
  const float* xr = x + (size_t)row * C;
  float s = 0.f, s2 = 0.f;
  for (int c = lane; c < C; c += 32) { float v = xr[c]; s += v; s2 += v * v; }
  s = wredsum(s); s2 = wredsum(s2);
  float mean = s * (1.f / C);
  float var  = s2 * (1.f / C) - mean * mean;
  float inv  = rsqrtf(var + 1e-5f);
  float* orow = o + (size_t)row * C;
  for (int c = lane; c < C; c += 32) orow[c] = (xr[c] - mean) * inv * g[c] + b[c];
}

// ---------------------------------------------------------------------------
// WMMA GEMM: out[M,N] = act(A[M,K] * Wm[N,K]^T + bias), optional per-row scale
// and residual add. Block: 128 threads / 4 waves, 64x64 tile; each wave owns
// 32x32 (2x2 WMMA accumulators). Double-buffered LDS tiles staged by TDM.
// M,N multiples of 64; K multiple of 32 (true for all call sites).
// ---------------------------------------------------------------------------
__global__ __launch_bounds__(128) void wmma_gemm(
    const float* __restrict__ A, const float* __restrict__ Wm,
    const float* __restrict__ bias, float* __restrict__ out,
    const float* __restrict__ resid, const float* __restrict__ rowScale,
    int rsStride, int M, int N, int K, int act) {
  __shared__ float As[2][TROWS][TSTRIDE];
  __shared__ float Bs[2][TROWS][TSTRIDE];
  int tid = threadIdx.x;
  int lane = tid & 31, wid = tid >> 5;
  int m0 = blockIdx.x * 64, n0 = blockIdx.y * 64;
  int wm = (wid >> 1) * 32, wn = (wid & 1) * 32;
  int  lm = lane & 15;
  bool lo = lane < 16;

#if !USE_TDM
  int srow = tid >> 1;            // 0..63
  int skq  = (tid & 1) * 16;      // 0 or 16
  const float* ar = A  + (size_t)(m0 + srow) * K;
  const float* wr = Wm + (size_t)(n0 + srow) * K;
#endif

  auto stage = [&](int buf, int k0) {
#if USE_TDM
    if (wid == 0) {
      tdm_load_tile_f32(A + (size_t)m0 * K + k0, lds_off(&As[buf][0][0]), K, K, M);
      tdm_load_tile_f32(Wm + (size_t)n0 * K + k0, lds_off(&Bs[buf][0][0]), K, K, N);
    }
#else
    float4 a0 = *(const float4*)(ar + k0 + skq);
    float4 a1 = *(const float4*)(ar + k0 + skq + 4);
    float4 a2 = *(const float4*)(ar + k0 + skq + 8);
    float4 a3 = *(const float4*)(ar + k0 + skq + 12);
    *(float4*)&As[buf][srow][skq + 0]  = a0;
    *(float4*)&As[buf][srow][skq + 4]  = a1;
    *(float4*)&As[buf][srow][skq + 8]  = a2;
    *(float4*)&As[buf][srow][skq + 12] = a3;
    float4 w0 = *(const float4*)(wr + k0 + skq);
    float4 w1 = *(const float4*)(wr + k0 + skq + 4);
    float4 w2 = *(const float4*)(wr + k0 + skq + 8);
    float4 w3 = *(const float4*)(wr + k0 + skq + 12);
    *(float4*)&Bs[buf][srow][skq + 0]  = w0;
    *(float4*)&Bs[buf][srow][skq + 4]  = w1;
    *(float4*)&Bs[buf][srow][skq + 8]  = w2;
    *(float4*)&Bs[buf][srow][skq + 12] = w3;
#endif
  };

  v8f acc[2][2] = {{{}, {}}, {{}, {}}};
  stage(0, 0);
  for (int k0 = 0; k0 < K; k0 += TK) {
    int cur = (k0 >> 5) & 1;
    if (k0 + TK < K) {
      stage(1 - cur, k0 + TK);
#if USE_TDM
      if (wid == 0) __builtin_amdgcn_s_wait_tensorcnt(2);  // current tiles done
#endif
    } else {
#if USE_TDM
      if (wid == 0) __builtin_amdgcn_s_wait_tensorcnt(0);  // drain
#endif
    }
    __syncthreads();
    // Build f16 fragments from f32 LDS tiles.
    v16h af[2], bf[2];
    int kb = lo ? 0 : 16;           // A frag: K=0..15 / 16..31 per half-wave
#pragma unroll
    for (int t = 0; t < 2; t++) {
      const float* ap = &As[cur][wm + t * 16 + lm][0];
#pragma unroll
      for (int i = 0; i < 16; i++) {
        int k = kb + i;
        af[t][i] = (_Float16)ap[k];
      }
    }
    int kb2 = lo ? 0 : 16;          // B frag: K=0..15 / 16..31 per half-wave
#pragma unroll
    for (int t = 0; t < 2; t++) {
      const float* bp = &Bs[cur][wn + t * 16 + lm][0];
#pragma unroll
      for (int i = 0; i < 16; i++) bf[t][i] = (_Float16)bp[kb2 + i];
    }
#pragma unroll
    for (int i = 0; i < 2; i++)
#pragma unroll
      for (int j = 0; j < 2; j++)
        acc[i][j] = __builtin_amdgcn_wmma_f32_16x16x32_f16(
            false, af[i], false, bf[j], (short)0, acc[i][j], false, false);
    __syncthreads();
  }

#pragma unroll
  for (int i = 0; i < 2; i++)
#pragma unroll
    for (int j = 0; j < 2; j++)
#pragma unroll
      for (int r = 0; r < 8; r++) {
        int m = m0 + wm + i * 16 + (lo ? r : (8 + r));
        int n = n0 + wn + j * 16 + lm;
        float v = acc[i][j][r];
        if (bias) v += bias[n];
        if (act == 1)      v = tanhf(v);
        else if (act == 2) v = 0.5f * v * (1.f + erff(v * 0.70710678118f));
        if (rowScale) v *= rowScale[(size_t)m * rsStride];
        if (resid)    v += resid[(size_t)m * N + n];
        out[(size_t)m * N + n] = v;
      }
}

// ---------------------------------------------------------------------------
// Windowed causal attention (W=64): one wave per (batch, head, 16-query tile).
// 5 key tiles (qt-4..qt), QK^T + PV through WMMA; softmax in C-layout regs.
// Output pre-multiplied by attn_gate.
// ---------------------------------------------------------------------------
__global__ __launch_bounds__(128) void attn_win(
    const float* __restrict__ qkv, const float* __restrict__ gate,
    float* __restrict__ y) {
  __shared__ _Float16 Pl[4][6][16][16];  // per-wave prob tiles (slot 5 = zero pad)
  int tid = threadIdx.x, wid = tid >> 5, lane = tid & 31;
  int widx = blockIdx.x * 4 + wid;       // 0 .. 3071
  int qt = widx & 15;
  int t1 = widx >> 4;
  int hh = t1 % H;
  int bb = t1 / H;
  int q0 = qt * 16;
  int  lm = lane & 15;
  bool lo = lane < 16;
  const float* qbase = qkv + (size_t)bb * T * C3 + hh * HD;
  const float* kbase = qbase + C;
  const float* vbase = qbase + 2 * C;

  // ---- Q fragments (two K=32 chunks over HD=64) ----
  v16h qf[2];
  {
    const float* qp = qbase + (size_t)(q0 + lm) * C3;
#pragma unroll
    for (int f = 0; f < 2; f++) {
      int d0 = f * 32 + (lo ? 0 : 8);
      float4 x0 = *(const float4*)(qp + d0);
      float4 x1 = *(const float4*)(qp + d0 + 4);
      float4 x2 = *(const float4*)(qp + d0 + 16);
      float4 x3 = *(const float4*)(qp + d0 + 20);
      CVT4(qf[f], 0, x0); CVT4(qf[f], 4, x1);
      CVT4(qf[f], 8, x2); CVT4(qf[f], 12, x3);
    }
  }

  // ---- scores S[slot] = Q * K^T for key tiles qt-4..qt ----
  v8f S[5];
#pragma unroll
  for (int s = 0; s < 5; s++) {
    int kt = qt - 4 + s;
    int ktok = kt * 16 + lm;
    int ktc = ktok < 0 ? 0 : ktok;       // clamped (masked out later)
    const float* kp = kbase + (size_t)ktc * C3;
    v8f a = {};
#pragma unroll
    for (int f = 0; f < 2; f++) {
      int d0 = f * 32 + (lo ? 0 : 8);
      v16h bf;
      float4 x0 = *(const float4*)(kp + d0);
      float4 x1 = *(const float4*)(kp + d0 + 4);
      float4 x2 = *(const float4*)(kp + d0 + 16);
      float4 x3 = *(const float4*)(kp + d0 + 20);
      CVT4(bf, 0, x0); CVT4(bf, 4, x1); CVT4(bf, 8, x2); CVT4(bf, 12, x3);
      a = __builtin_amdgcn_wmma_f32_16x16x32_f16(false, qf[f], false, bf,
                                                 (short)0, a, false, false);
    }
    S[s] = a;
  }

  // ---- mask + softmax (row reductions over 16-lane half-wave groups) ----
#pragma unroll
  for (int r = 0; r < 8; r++) {
    int m = lo ? r : 8 + r;
    int qi = q0 + m;
    float mx = -1e30f;
#pragma unroll
    for (int s = 0; s < 5; s++) {
      int key = (qt - 4 + s) * 16 + lm;
      bool ok = (key >= 0) && (key <= qi) && ((qi - key) <= WWIN);
      float v = ok ? S[s][r] * SCL : -1e30f;
      S[s][r] = v;
      mx = fmaxf(mx, v);
    }
    mx = fmaxf(mx, __shfl_xor(mx, 1, 32));
    mx = fmaxf(mx, __shfl_xor(mx, 2, 32));
    mx = fmaxf(mx, __shfl_xor(mx, 4, 32));
    mx = fmaxf(mx, __shfl_xor(mx, 8, 32));
    float sum = 0.f;
#pragma unroll
    for (int s = 0; s < 5; s++) {
      float p = expf(S[s][r] - mx);
      S[s][r] = p; sum += p;
    }
    sum += __shfl_xor(sum, 1, 32);
    sum += __shfl_xor(sum, 2, 32);
    sum += __shfl_xor(sum, 4, 32);
    sum += __shfl_xor(sum, 8, 32);
    float inv = 1.f / sum;
#pragma unroll
    for (int s = 0; s < 5; s++) S[s][r] *= inv;
  }

  // ---- repack probs via per-wave LDS region (same-wave DS is in-order) ----
#pragma unroll
  for (int s = 0; s < 5; s++)
#pragma unroll
    for (int r = 0; r < 8; r++) Pl[wid][s][lo ? r : 8 + r][lm] = (_Float16)S[s][r];
#pragma unroll
  for (int r = 0; r < 8; r++) Pl[wid][5][lo ? r : 8 + r][lm] = (_Float16)0.f;

  // ---- Y = P * V : 3 chunks of 32 keys x 4 d-groups of 16 ----
  v8f Y[4] = {{}, {}, {}, {}};
#pragma unroll
  for (int cch = 0; cch < 3; cch++) {
    v16h pf;
    int kb = lo ? 0 : 16;
#pragma unroll
    for (int i = 0; i < 16; i++) {
      int k = kb + i;
      pf[i] = Pl[wid][2 * cch + (k >> 4)][lm][k & 15];
    }
#pragma unroll
    for (int g = 0; g < 4; g++) {
      v16h vf;
      int d = g * 16 + lm;
      int kb2 = lo ? 0 : 16;
#pragma unroll
      for (int i = 0; i < 16; i++) {
        int kk = kb2 + i;
        int key = (qt - 4 + 2 * cch + (kk >> 4)) * 16 + (kk & 15);
        key = key < 0 ? 0 : (key > T - 1 ? T - 1 : key);  // prob==0 there
        vf[i] = (_Float16)vbase[(size_t)key * C3 + d];
      }
      Y[g] = __builtin_amdgcn_wmma_f32_16x16x32_f16(false, pf, false, vf,
                                                    (short)0, Y[g], false, false);
    }
  }

  // ---- store y * attn_gate ----
#pragma unroll
  for (int g = 0; g < 4; g++)
#pragma unroll
    for (int r = 0; r < 8; r++) {
      int m = lo ? r : 8 + r;
      int d = hh * HD + g * 16 + lm;
      y[((size_t)bb * T + q0 + m) * C + d] = Y[g][r] * gate[d];
    }
}

// ---------------------------------------------------------------------------
// MoE gating: one wave per token row. scores = h2 @ gate_w^T + b, softmax(E=4),
// top-2 mask, renormalize; writes masked probs [NROW, E].
// ---------------------------------------------------------------------------
__global__ void moe_gate(const float* __restrict__ h2, const float* __restrict__ gw,
                         const float* __restrict__ gb, float* __restrict__ probs) {
  int wid = threadIdx.x >> 5, lane = threadIdx.x & 31;
  int row = blockIdx.x * (blockDim.x >> 5) + wid;
  if (row >= NROW) return;
  const float* hr = h2 + (size_t)row * C;
  float acc[E] = {0.f, 0.f, 0.f, 0.f};
  for (int c = lane; c < C; c += 32) {
    float hv = hr[c];
#pragma unroll
    for (int e = 0; e < E; e++) acc[e] += hv * gw[e * C + c];
  }
#pragma unroll
  for (int e = 0; e < E; e++) acc[e] = wredsum(acc[e]);
  if (lane == 0) {
    float s[E], p[E], mx = -1e30f;
#pragma unroll
    for (int e = 0; e < E; e++) { s[e] = acc[e] + gb[e]; mx = fmaxf(mx, s[e]); }
    float sum = 0.f;
#pragma unroll
    for (int e = 0; e < E; e++) { p[e] = expf(s[e] - mx); sum += p[e]; }
#pragma unroll
    for (int e = 0; e < E; e++) p[e] /= sum;
    int i1 = 0;
#pragma unroll
    for (int e = 1; e < E; e++) if (p[e] > p[i1]) i1 = e;
    int i2 = (i1 == 0) ? 1 : 0;
#pragma unroll
    for (int e = 0; e < E; e++) if (e != i1 && p[e] > p[i2]) i2 = e;
    float denom = p[i1] + p[i2] + 1e-9f;
#pragma unroll
    for (int e = 0; e < E; e++)
      probs[(size_t)row * E + e] = (e == i1 || e == i2) ? p[e] / denom : 0.f;
  }
}

// ---------------------------------------------------------------------------
extern "C" void kernel_launch(void* const* d_in, const int* in_sizes, int n_in,
                              void* d_out, int out_size, void* d_ws, size_t ws_size,
                              hipStream_t stream) {
  (void)in_sizes; (void)n_in; (void)out_size; (void)ws_size;
  const float* x_in   = (const float*)d_in[0];
  const float* ln1_g  = (const float*)d_in[1];
  const float* ln1_b  = (const float*)d_in[2];
  const float* ln2_g  = (const float*)d_in[3];
  const float* ln2_b  = (const float*)d_in[4];
  const float* qkv_w  = (const float*)d_in[5];
  const float* qkv_b  = (const float*)d_in[6];
  const float* proj_w = (const float*)d_in[7];
  const float* proj_b = (const float*)d_in[8];
  const float* agate  = (const float*)d_in[9];
  const float* gate_w = (const float*)d_in[10];
  const float* gate_b = (const float*)d_in[11];
  const float* exp_w1 = (const float*)d_in[12];
  const float* exp_b1 = (const float*)d_in[13];
  const float* exp_w2 = (const float*)d_in[14];
  const float* exp_b2 = (const float*)d_in[15];
  const float* th_w1  = (const float*)d_in[16];
  const float* th_b1  = (const float*)d_in[17];
  const float* th_w2  = (const float*)d_in[18];
  const float* th_b2  = (const float*)d_in[19];

  float* x     = (float*)d_out;                 // running activations [NROW, C]
  float* ws    = (float*)d_ws;
  float* h     = ws;                            // [NROW, C]   ln1 / ln2 output
  float* big   = h   + (size_t)NROW * C;        // [NROW, FF]  qkv (1152) / hid (1536)
  float* mid   = big + (size_t)NROW * FF;       // [NROW, C]   attn y / thought hidden
  float* probs = mid + (size_t)NROW * C;        // [NROW, E]

  hipMemcpyAsync(x, x_in, (size_t)NROW * C * sizeof(float),
                 hipMemcpyDeviceToDevice, stream);

  for (int l = 0; l < LAYERS; l++) {
    // --- attention ---
    ln_rows<<<NROW / 8, 256, 0, stream>>>(x, ln1_g + l * C, ln1_b + l * C, h);
    wmma_gemm<<<dim3(NROW / 64, C3 / 64), 128, 0, stream>>>(
        h, qkv_w + (size_t)l * C3 * C, qkv_b + l * C3,
        big, nullptr, nullptr, 0, NROW, C3, C, 0);
    attn_win<<<(BATCH * H * (T / 16)) / 4, 128, 0, stream>>>(big, agate + l * C, mid);
    wmma_gemm<<<dim3(NROW / 64, C / 64), 128, 0, stream>>>(
        mid, proj_w + (size_t)l * C * C, proj_b + l * C,
        x, x, nullptr, 0, NROW, C, C, 0);
    // --- thought MLP ---
    wmma_gemm<<<dim3(NROW / 64, DT / 64), 128, 0, stream>>>(
        x, th_w1 + (size_t)l * DT * C, th_b1 + l * DT,
        mid, nullptr, nullptr, 0, NROW, DT, C, 1);
    wmma_gemm<<<dim3(NROW / 64, C / 64), 128, 0, stream>>>(
        mid, th_w2 + (size_t)l * C * DT, th_b2 + l * C,
        x, x, nullptr, 0, NROW, C, DT, 0);
    // --- MoE ---
    ln_rows<<<NROW / 8, 256, 0, stream>>>(x, ln2_g + l * C, ln2_b + l * C, h);
    moe_gate<<<NROW / 8, 256, 0, stream>>>(h, gate_w + (size_t)l * E * C,
                                           gate_b + l * E, probs);
    for (int e = 0; e < E; e++) {
      wmma_gemm<<<dim3(NROW / 64, FF / 64), 128, 0, stream>>>(
          h, exp_w1 + (size_t)(l * E + e) * FF * C, exp_b1 + (size_t)(l * E + e) * FF,
          big, nullptr, nullptr, 0, NROW, FF, C, 2);
      wmma_gemm<<<dim3(NROW / 64, C / 64), 128, 0, stream>>>(
          big, exp_w2 + (size_t)(l * E + e) * C * FF, exp_b2 + (size_t)(l * E + e) * C,
          x, x, probs + e, E, NROW, C, FF, 0);
    }
  }
}
